// Block_74337293959216
// MI455X (gfx1250) — matrix operand order repaired and checked by reading, not compiled, not run on hardware
//
#include <hip/hip_runtime.h>

#define BATCH 2
#define SEQ   2048
#define DEMB  1024
#define NHEAD 16
#define HSZ   64
#define BT    (BATCH * SEQ)

typedef __attribute__((ext_vector_type(16))) __bf16       bf16x16;
typedef __attribute__((ext_vector_type(8)))  float        f32x8;
typedef __attribute__((ext_vector_type(4)))  unsigned int u32x4;

union Frag {
    bf16x16 v;
    u32x4   q[2];
};

__device__ __forceinline__ unsigned short f2bf(float f) {
    unsigned int u = __float_as_uint(f);
    u += 0x7FFFu + ((u >> 16) & 1u);          // round-to-nearest-even
    return (unsigned short)(u >> 16);
}
__device__ __forceinline__ float bf2f(unsigned short h) {
    return __uint_as_float(((unsigned int)h) << 16);
}
__device__ __forceinline__ f32x8 zero8() {
    f32x8 z;
#pragma unroll
    for (int i = 0; i < 8; ++i) z[i] = 0.0f;
    return z;
}
__device__ __forceinline__ f32x8 wmma_bf16(Frag a, Frag b, f32x8 c) {
    return __builtin_amdgcn_wmma_f32_16x16x32_bf16(false, a.v, false, b.v,
                                                   (short)0, c, false, false);
}

// ---------------------------------------------------------------------------
// LayerNorm: f32 [rows, DEMB] -> bf16 [rows, DEMB]
// ---------------------------------------------------------------------------
__global__ __launch_bounds__(256) void ln_kernel(
    const float* __restrict__ x, const float* __restrict__ g,
    const float* __restrict__ bta, unsigned short* __restrict__ out)
{
    __shared__ float red[256];
    const int row = blockIdx.x;
    const int tid = threadIdx.x;
    const float* xr = x + (size_t)row * DEMB;
    float s = 0.f, s2 = 0.f;
    for (int i = tid; i < DEMB; i += 256) { float v = xr[i]; s += v; s2 += v * v; }
    red[tid] = s; __syncthreads();
    for (int o2 = 128; o2 > 0; o2 >>= 1) {
        if (tid < o2) red[tid] += red[tid + o2];
        __syncthreads();
    }
    const float mean = red[0] * (1.0f / DEMB);
    __syncthreads();
    red[tid] = s2; __syncthreads();
    for (int o2 = 128; o2 > 0; o2 >>= 1) {
        if (tid < o2) red[tid] += red[tid + o2];
        __syncthreads();
    }
    const float var  = red[0] * (1.0f / DEMB) - mean * mean;
    const float rstd = rsqrtf(var + 1e-5f);
    unsigned short* orow = out + (size_t)row * DEMB;
    for (int i = tid; i < DEMB; i += 256)
        orow[i] = f2bf((xr[i] - mean) * rstd * g[i] + bta[i]);
}

// ---------------------------------------------------------------------------
// Weight packing into WMMA-B fragment order.
// Bp layout: group = (n/16)*(K/32) + (k/32); flat[(group*32 + lane)*16 + e]
// with n = base_n + lane%16, k = base_k + 16*(lane/16) + e.
// ---------------------------------------------------------------------------
__global__ __launch_bounds__(256) void pack_rm_kernel(
    const float* __restrict__ W, unsigned short* __restrict__ Bp, int K, int N)
{
    const int tid = blockIdx.x * 256 + threadIdx.x;
    const int KT = K >> 5;
    const int total = (N >> 4) * KT * 32;
    if (tid >= total) return;
    const int lane = tid & 31;
    const int kt = (tid >> 5) % KT;
    const int nt = (tid >> 5) / KT;
    const int n  = nt * 16 + (lane & 15);
    const int kb = kt * 32 + (lane >> 4) * 16;
    unsigned short* dst = Bp + (size_t)tid * 16;
#pragma unroll
    for (int e = 0; e < 16; ++e)
        dst[e] = f2bf(W[(size_t)(kb + e) * N + n]);
}

// Wq/Wk/Wv are [H, D, HS]; packed as a single [D x 3D] B matrix.
__global__ __launch_bounds__(256) void pack_qkv_kernel(
    const float* __restrict__ Wq, const float* __restrict__ Wk,
    const float* __restrict__ Wv, unsigned short* __restrict__ Bp)
{
    const int tid = blockIdx.x * 256 + threadIdx.x;
    const int KT = DEMB >> 5;
    const int total = ((3 * DEMB) >> 4) * KT * 32;
    if (tid >= total) return;
    const int lane = tid & 31;
    const int kt = (tid >> 5) % KT;
    const int nt = (tid >> 5) / KT;
    const int n  = nt * 16 + (lane & 15);
    const int kb = kt * 32 + (lane >> 4) * 16;
    const int sel = n >> 10;
    const int c   = n & (DEMB - 1);
    const int h = c >> 6, hs = c & 63;
    const float* W = (sel == 0) ? Wq : ((sel == 1) ? Wk : Wv);
    unsigned short* dst = Bp + (size_t)tid * 16;
#pragma unroll
    for (int e = 0; e < 16; ++e)
        dst[e] = f2bf(W[((size_t)h * DEMB + (kb + e)) * HSZ + hs]);
}

// ---------------------------------------------------------------------------
// Generic bf16 WMMA GEMM: C[M,N] = act(A[M,K] @ Bp (+bias) (+res))
// block = 256 threads = 8 waves (2 M x 4 N), wave tile 64x64 (4x4 WMMA),
// block tile 128 x 256, K-step 32. 16 b128 loads feed 16 WMMAs per step.
// ---------------------------------------------------------------------------
__global__ __launch_bounds__(256) void gemm_bf16_kernel(
    const unsigned short* __restrict__ A,
    const unsigned short* __restrict__ Bp,
    const float* __restrict__ bias,
    const float* __restrict__ res,
    float* __restrict__ outf,
    unsigned short* __restrict__ outb,
    int M, int N, int K, int relu)
{
    const int lane = threadIdx.x & 31;
    const int wave = threadIdx.x >> 5;
    const int l15  = lane & 15;
    const int half = lane >> 4;
    const int KT   = K >> 5;
    const int m0 = blockIdx.y * 128 + (wave >> 2) * 64;
    const int n0 = blockIdx.x * 256 + (wave & 3) * 64;

    f32x8 acc[4][4];
#pragma unroll
    for (int i = 0; i < 4; ++i)
#pragma unroll
        for (int j = 0; j < 4; ++j) acc[i][j] = zero8();

    const unsigned short* arow[4];
#pragma unroll
    for (int mi = 0; mi < 4; ++mi)
        arow[mi] = A + (size_t)(m0 + mi * 16 + l15) * K + half * 16;
    const unsigned short* bpt[4];
#pragma unroll
    for (int ni = 0; ni < 4; ++ni)
        bpt[ni] = Bp + ((size_t)(n0 >> 4) + ni) * KT * 512 + lane * 16;

    for (int k0 = 0; k0 < K; k0 += 32) {
        Frag a[4], b[4];
#pragma unroll
        for (int mi = 0; mi < 4; ++mi) {
            a[mi].q[0] = *(const u32x4*)(arow[mi] + k0);
            a[mi].q[1] = *(const u32x4*)(arow[mi] + k0 + 8);
        }
#pragma unroll
        for (int ni = 0; ni < 4; ++ni) {
            const unsigned short* g = bpt[ni] + (size_t)(k0 >> 5) * 512;
            b[ni].q[0] = *(const u32x4*)(g);
            b[ni].q[1] = *(const u32x4*)(g + 8);
        }
        // prefetch 4 K-steps ahead, once per 128-wide K chunk
        if (((k0 & 127) == 0) && (k0 + 128 < K)) {
#pragma unroll
            for (int mi = 0; mi < 4; ++mi)
                __builtin_prefetch(arow[mi] + k0 + 128, 0, 0);
#pragma unroll
            for (int ni = 0; ni < 4; ++ni)
                __builtin_prefetch(bpt[ni] + ((size_t)(k0 >> 5) + 4) * 512, 0, 0);
        }
#pragma unroll
        for (int mi = 0; mi < 4; ++mi)
#pragma unroll
            for (int ni = 0; ni < 4; ++ni)
                acc[mi][ni] = wmma_bf16(a[mi], b[ni], acc[mi][ni]);
    }

#pragma unroll
    for (int mi = 0; mi < 4; ++mi)
#pragma unroll
        for (int ni = 0; ni < 4; ++ni)
#pragma unroll
            for (int r = 0; r < 8; ++r) {
                const int m = m0 + mi * 16 + r + half * 8;
                const int n = n0 + ni * 16 + l15;
                float v = acc[mi][ni][r];
                if (bias) v += bias[n];
                if (res)  v += res[(size_t)m * N + n];
                if (relu) v = fmaxf(v, 0.0f);
                const size_t o = (size_t)m * N + n;
                if (outf) outf[o] = v;
                if (outb) outb[o] = f2bf(v);
            }
}

// ---------------------------------------------------------------------------
// Scatter fused QKV [BT, 3D] bf16 -> Q [bh,T,64] (x 1/sqrt(HS)),
// K [bh,T,64], Vt [bh,64,T]
// ---------------------------------------------------------------------------
__global__ __launch_bounds__(256) void scatter_qkv_kernel(
    const unsigned short* __restrict__ qkv,
    unsigned short* __restrict__ Q, unsigned short* __restrict__ Kd,
    unsigned short* __restrict__ Vt)
{
    const int idx = blockIdx.x * 256 + threadIdx.x;     // BT*DEMB threads
    const int c   = idx & (DEMB - 1);
    const int row = idx >> 10;
    const int b = row / SEQ, t = row % SEQ;
    const int h = c >> 6, hs = c & 63;
    const unsigned short* src = qkv + (size_t)row * (3 * DEMB) + c;
    const int bh = b * NHEAD + h;
    Q [((size_t)bh * SEQ + t) * HSZ + hs] = f2bf(bf2f(src[0]) * 0.125f);
    Kd[((size_t)bh * SEQ + t) * HSZ + hs] = src[DEMB];
    Vt[((size_t)bh * HSZ + hs) * SEQ + t] = src[2 * DEMB];
}

// ---------------------------------------------------------------------------
// Flash attention: 128 threads = 4 waves, each wave owns a 16-query tile.
// ---------------------------------------------------------------------------
__global__ __launch_bounds__(128) void attn_kernel(
    const unsigned short* __restrict__ Q,
    const unsigned short* __restrict__ Km,
    const unsigned short* __restrict__ Vt,
    unsigned short* __restrict__ att)
{
    __shared__ __align__(16) unsigned short plds[4][16][32];
    const int lane = threadIdx.x & 31;
    const int wave = threadIdx.x >> 5;
    const int l15  = lane & 15;
    const int half = lane >> 4;
    const int bh = blockIdx.x >> 5;       // 32 query-groups per (b,h)
    const int qg = blockIdx.x & 31;
    const int q0 = (qg * 4 + wave) * 16;
    const int b  = bh / NHEAD;
    const int h  = bh % NHEAD;

    const unsigned short* Qb = Q  + (size_t)bh * SEQ * HSZ;
    const unsigned short* Kb = Km + (size_t)bh * SEQ * HSZ;
    const unsigned short* Vb = Vt + (size_t)bh * HSZ * SEQ;

    Frag aq[2];
    {
        const unsigned short* qr = Qb + (size_t)(q0 + l15) * HSZ;
        aq[0].q[0] = *(const u32x4*)(qr + half * 16);
        aq[0].q[1] = *(const u32x4*)(qr + 8 + half * 16);
        aq[1].q[0] = *(const u32x4*)(qr + 32 + half * 16);
        aq[1].q[1] = *(const u32x4*)(qr + 40 + half * 16);
    }

    f32x8 o[4];
#pragma unroll
    for (int n = 0; n < 4; ++n) o[n] = zero8();
    float runmax[8], runsum[8];
#pragma unroll
    for (int r = 0; r < 8; ++r) { runmax[r] = -1e30f; runsum[r] = 0.0f; }

    for (int s0 = 0; s0 < q0 + 16; s0 += 32) {
        f32x8 sL = zero8(), sR = zero8();
#pragma unroll
        for (int c = 0; c < 2; ++c) {   // hs chunks 0..31, 32..63
            const unsigned short* kr = Kb + (size_t)(s0 + l15) * HSZ + c * 32 + half * 16;
            Frag bk;
            bk.q[0] = *(const u32x4*)kr;
            bk.q[1] = *(const u32x4*)(kr + 8);
            sL = wmma_bf16(aq[c], bk, sL);
            const unsigned short* kr2 = kr + 16 * HSZ;   // keys s0+16..s0+31
            bk.q[0] = *(const u32x4*)kr2;
            bk.q[1] = *(const u32x4*)(kr2 + 8);
            sR = wmma_bf16(aq[c], bk, sR);
        }
        // causal mask + online softmax (rows live per 16-lane half at vgpr r)
#pragma unroll
        for (int r = 0; r < 8; ++r) {
            const int m = q0 + r + half * 8;
            float vl = (s0 + l15      <= m) ? sL[r] : -1e30f;
            float vr = (s0 + 16 + l15 <= m) ? sR[r] : -1e30f;
            float mx = fmaxf(vl, vr);
#pragma unroll
            for (int d = 1; d < 16; d <<= 1) mx = fmaxf(mx, __shfl_xor(mx, d, 32));
            const float nm    = fmaxf(runmax[r], mx);
            const float alpha = __expf(runmax[r] - nm);
            runmax[r] = nm;
            const float pl = __expf(vl - nm);
            const float pr = __expf(vr - nm);
            float ps = pl + pr;
#pragma unroll
            for (int d = 1; d < 16; d <<= 1) ps += __shfl_xor(ps, d, 32);
            runsum[r] = runsum[r] * alpha + ps;
#pragma unroll
            for (int n = 0; n < 4; ++n) o[n][r] *= alpha;
            plds[wave][r + half * 8][l15]      = f2bf(pl);
            plds[wave][r + half * 8][l15 + 16] = f2bf(pr);
        }
        // reload P as an A fragment (same-wave LDS is in-order)
        Frag ap;
        ap.q[0] = *(const u32x4*)&plds[wave][l15][half * 16];
        ap.q[1] = *(const u32x4*)&plds[wave][l15][half * 16 + 8];
        // O += P @ V  (V transposed -> contiguous B fragments)
#pragma unroll
        for (int n = 0; n < 4; ++n) {
            const unsigned short* vr2 = Vb + (size_t)(n * 16 + l15) * SEQ + s0 + half * 16;
            Frag bv;
            bv.q[0] = *(const u32x4*)vr2;
            bv.q[1] = *(const u32x4*)(vr2 + 8);
            o[n] = wmma_bf16(ap, bv, o[n]);
        }
    }
#pragma unroll
    for (int n = 0; n < 4; ++n)
#pragma unroll
        for (int r = 0; r < 8; ++r) {
            const int t  = q0 + r + half * 8;
            const int hs = n * 16 + l15;
            const float val = o[n][r] / runsum[r];
            att[((size_t)(b * SEQ + t)) * DEMB + h * HSZ + hs] = f2bf(val);
        }
}

// ---------------------------------------------------------------------------
extern "C" void kernel_launch(void* const* d_in, const int* in_sizes, int n_in,
                              void* d_out, int out_size, void* d_ws, size_t ws_size,
                              hipStream_t stream) {
    (void)in_sizes; (void)n_in; (void)out_size; (void)ws_size;
    const float* x     = (const float*)d_in[0];
    const float* ln1_g = (const float*)d_in[1];
    const float* ln1_b = (const float*)d_in[2];
    const float* Wq    = (const float*)d_in[3];
    const float* Wk    = (const float*)d_in[4];
    const float* Wv    = (const float*)d_in[5];
    const float* Wo    = (const float*)d_in[6];
    const float* bo    = (const float*)d_in[7];
    const float* ln2_g = (const float*)d_in[8];
    const float* ln2_b = (const float*)d_in[9];
    const float* W1    = (const float*)d_in[10];
    const float* b1    = (const float*)d_in[11];
    const float* W2    = (const float*)d_in[12];
    const float* b2    = (const float*)d_in[13];

    char* ws = (char*)d_ws;
    size_t off = 0;
    auto alloc = [&](size_t bytes) -> void* {
        void* p = ws + off;
        off = (off + bytes + 255) & ~(size_t)255;
        return p;
    };
    unsigned short* qkvp = (unsigned short*)alloc((size_t)3 * DEMB * DEMB * 2);
    unsigned short* Wop  = (unsigned short*)alloc((size_t)DEMB * DEMB * 2);
    unsigned short* W1p  = (unsigned short*)alloc((size_t)DEMB * 4 * DEMB * 2);
    unsigned short* W2p  = (unsigned short*)alloc((size_t)4 * DEMB * DEMB * 2);
    unsigned short* xn   = (unsigned short*)alloc((size_t)BT * DEMB * 2);
    unsigned short* qkv  = (unsigned short*)alloc((size_t)BT * 3 * DEMB * 2);
    unsigned short* Qb   = (unsigned short*)alloc((size_t)BATCH * NHEAD * SEQ * HSZ * 2);
    unsigned short* Kb   = (unsigned short*)alloc((size_t)BATCH * NHEAD * SEQ * HSZ * 2);
    unsigned short* Vtb  = (unsigned short*)alloc((size_t)BATCH * NHEAD * HSZ * SEQ * 2);
    unsigned short* attb = (unsigned short*)alloc((size_t)BT * DEMB * 2);
    float*          x1   = (float*)         alloc((size_t)BT * DEMB * 4);
    unsigned short* xn2  = (unsigned short*)alloc((size_t)BT * DEMB * 2);
    unsigned short* hbuf = (unsigned short*)alloc((size_t)BT * 4 * DEMB * 2);

    // --- weight packing (bf16, B-fragment order) ---
    pack_qkv_kernel<<<768, 256, 0, stream>>>(Wq, Wk, Wv, qkvp);
    pack_rm_kernel<<<256,  256, 0, stream>>>(Wo, Wop, DEMB, DEMB);
    pack_rm_kernel<<<1024, 256, 0, stream>>>(W1, W1p, DEMB, 4 * DEMB);
    pack_rm_kernel<<<1024, 256, 0, stream>>>(W2, W2p, 4 * DEMB, DEMB);

    // --- ln1 -> xn (bf16) ---
    ln_kernel<<<BT, 256, 0, stream>>>(x, ln1_g, ln1_b, xn);

    // --- fused QKV GEMM: [BT,1024] x [1024,3072] -> bf16 ---
    gemm_bf16_kernel<<<dim3(3 * DEMB / 256, BT / 128), 256, 0, stream>>>(
        xn, qkvp, nullptr, nullptr, nullptr, qkv, BT, 3 * DEMB, DEMB, 0);

    // --- scatter Q/K/Vt (fold 1/sqrt(HS) into Q) ---
    scatter_qkv_kernel<<<(BT * DEMB) / 256, 256, 0, stream>>>(qkv, Qb, Kb, Vtb);

    // --- flash attention -> attb (bf16, heads concatenated) ---
    attn_kernel<<<BATCH * NHEAD * (SEQ / 64), 128, 0, stream>>>(Qb, Kb, Vtb, attb);

    // --- out-proj + bias + residual(x) -> x1 (f32) ---
    gemm_bf16_kernel<<<dim3(DEMB / 256, BT / 128), 256, 0, stream>>>(
        attb, Wop, bo, x, x1, nullptr, BT, DEMB, DEMB, 0);

    // --- ln2 -> xn2 (bf16) ---
    ln_kernel<<<BT, 256, 0, stream>>>(x1, ln2_g, ln2_b, xn2);

    // --- MLP1: relu(xn2 @ W1 + b1) -> h (bf16) ---
    gemm_bf16_kernel<<<dim3(4 * DEMB / 256, BT / 128), 256, 0, stream>>>(
        xn2, W1p, b1, nullptr, nullptr, hbuf, BT, 4 * DEMB, DEMB, 1);

    // --- MLP2: x1 + h @ W2 + b2 -> d_out (f32) ---
    gemm_bf16_kernel<<<dim3(DEMB / 256, BT / 128), 256, 0, stream>>>(
        hbuf, W2p, b2, x1, (float*)d_out, nullptr, BT, DEMB, 4 * DEMB, 0);
}
